// MAB_qkv_22557168238600
// MI455X (gfx1250) — matrix operand order repaired and checked by reading, compiled, not run on hardware
//
#include <hip/hip_runtime.h>
#include <cstdint>

typedef __attribute__((ext_vector_type(16))) __bf16 v16bf;
typedef __attribute__((ext_vector_type(8)))  __bf16 v8bf;
typedef __attribute__((ext_vector_type(8)))  unsigned short v8us;
typedef __attribute__((ext_vector_type(8)))  float  v8f;
typedef __attribute__((ext_vector_type(4)))  unsigned v4u;
typedef __attribute__((ext_vector_type(8)))  unsigned v8u;

union FragBF {
  v16bf v;
  struct { uint4 lo; uint4 hi; } q;
  __bf16 e[16];
};

// A-fragment (16x32, M x K), source row-major [m][k] with given pitch.
// Layout used here (same k order as frag_b): lane m=lane&15, h=lane>>4; VGPR0-7 hold K=16h..16h+15 (contiguous).
static __device__ __forceinline__ v16bf frag_a(const __bf16* base, int row0, int pitch, int lane) {
  const int m = lane & 15, h = lane >> 4;
  const __bf16* p = base + (size_t)(row0 + m) * pitch;
  FragBF f;
  f.q.lo = *(const uint4*)(p + 16 * h);
  f.q.hi = *(const uint4*)(p + 16 * h + 8);
  return f.v;
}

// B-fragment (32x16, K x N), source stored as B^T row-major [n][k] with given pitch.
// Layout used here (same k order as frag_a): lane n=lane&15, h=lane>>4; VGPR0-7 hold K=16h..16h+15 (contiguous).
static __device__ __forceinline__ v16bf frag_b(const __bf16* base, int row0, int pitch, int lane) {
  const int n = lane & 15, h = lane >> 4;
  const __bf16* p = base + (size_t)(row0 + n) * pitch + 16 * h;
  FragBF f;
  f.q.lo = *(const uint4*)(p);
  f.q.hi = *(const uint4*)(p + 8);
  return f.v;
}

static __device__ __forceinline__ v8bf cvt8(const float* p) {
  float4 f0 = *(const float4*)p;
  float4 f1 = *(const float4*)(p + 4);
  v8bf o;
  o[0] = (__bf16)f0.x; o[1] = (__bf16)f0.y; o[2] = (__bf16)f0.z; o[3] = (__bf16)f0.w;
  o[4] = (__bf16)f1.x; o[5] = (__bf16)f1.y; o[6] = (__bf16)f1.z; o[7] = (__bf16)f1.w;
  return o;
}

// TDM: DMA a 32-row x 512-col bf16 tile (contiguous rows) from global to LDS.
// D# per CDNA5 ISA 8.3/8.4: group0 {count=1, lds_addr, global_addr, type=2},
// group1 {data_size=2B, tensor_dim0=512, tensor_dim1=32, tile 512x32, stride0=512}.
static __device__ __forceinline__ void tdm_load_k32x512(const __bf16* gptr, unsigned lds_off) {
  const unsigned long long ga = (unsigned long long)(uintptr_t)gptr;
  v4u g0;
  g0[0] = 1u;                                                  // count=1 (valid user D#)
  g0[1] = lds_off;                                             // lds_addr
  g0[2] = (unsigned)ga;                                        // global_addr[31:0]
  g0[3] = (unsigned)((ga >> 32) & 0x01ffffffu) | (2u << 30);   // global_addr[56:32], type=2
  v8u g1;
  g1[0] = 0x00010000u;   // workgroup_mask=0, data_size=1 (2 bytes)
  g1[1] = 512u << 16;    // tensor_dim0 = 512 (bits 79:48)
  g1[2] = 32u << 16;     // tensor_dim1 = 32  (bits 111:80)
  g1[3] = 512u << 16;    // tile_dim0 = 512   (bits 127:112)
  g1[4] = 32u;           // tile_dim1 = 32    (bits 143:128)
  g1[5] = 512u;          // tensor_dim0_stride = 512 (bits 207:160)
  g1[6] = 0u;
  g1[7] = 0u;
  asm volatile("tensor_load_to_lds %0, %1" :: "s"(g0), "s"(g1) : "memory");
}

// ---------------------------------------------------------------------------
// GEMM: Out[M][N] = X[M][K] @ W[N][K]^T + bias   (torch Linear convention)
// Tile 64x64, k-slice 64 -> 4 WMMAs per wave per barrier pair.
// bf16 input path stages A via async global->LDS (ASYNCcnt).
// ---------------------------------------------------------------------------
template <bool IN_BF16, bool OUT_F32>
__global__ void __launch_bounds__(256)
proj_kernel(const void* __restrict__ Xin, const float* __restrict__ W,
            const float* __restrict__ bias, void* __restrict__ Outp,
            int M, int K, int N) {
  __shared__ __bf16 As[64 * 64];
  __shared__ __bf16 Bs[64 * 64];

  const int tid  = threadIdx.x;
  const int lane = tid & 31;
  const int wave = tid >> 5;
  const int mi2  = wave & 3;   // 16-row subtile (0..3)
  const int nj2  = wave >> 2;  // 32-col subtile (0..1)
  const int rowBase = blockIdx.x * 64;
  const int nBase   = blockIdx.y * 64;

  v8f acc0 = {0, 0, 0, 0, 0, 0, 0, 0};
  v8f acc1 = {0, 0, 0, 0, 0, 0, 0, 0};

  const int lr = tid >> 2;         // 0..63
  const int lc = (tid & 3) * 16;   // 0,16,32,48

  for (int kk = 0; kk < K; kk += 64) {
    if constexpr (IN_BF16) {
      const __bf16* X = (const __bf16*)Xin;
      const __bf16* gsrc = &X[(size_t)(rowBase + lr) * K + kk + lc];
      const unsigned lds_off = (unsigned)(uintptr_t)&As[lr * 64 + lc];
      const unsigned long long gaddr = (unsigned long long)(uintptr_t)gsrc;
      asm volatile("global_load_async_to_lds_b128 %0, %1, off"
                   :: "v"(lds_off), "v"(gaddr) : "memory");
      asm volatile("global_load_async_to_lds_b128 %0, %1, off offset:16"
                   :: "v"(lds_off), "v"(gaddr) : "memory");
    } else {
      const float* X = (const float*)Xin;
      const float* xp = &X[(size_t)(rowBase + lr) * K + kk + lc];
      *(v8bf*)&As[lr * 64 + lc]     = cvt8(xp);
      *(v8bf*)&As[lr * 64 + lc + 8] = cvt8(xp + 8);
    }
    {
      const float* wp = &W[(size_t)(nBase + lr) * K + kk + lc];
      *(v8bf*)&Bs[lr * 64 + lc]     = cvt8(wp);
      *(v8bf*)&Bs[lr * 64 + lc + 8] = cvt8(wp + 8);
    }
    if constexpr (IN_BF16)
      asm volatile("s_wait_asynccnt 0x0" ::: "memory");
    __syncthreads();
#pragma unroll
    for (int ks = 0; ks < 64; ks += 32) {
      v16bf a  = frag_a(As + ks, 16 * mi2, 64, lane);
      v16bf b0 = frag_b(Bs + ks, 32 * nj2, 64, lane);
      v16bf b1 = frag_b(Bs + ks, 32 * nj2 + 16, 64, lane);
      acc0 = __builtin_amdgcn_wmma_f32_16x16x32_bf16(false, a, false, b0, (short)0, acc0, false, false);
      acc1 = __builtin_amdgcn_wmma_f32_16x16x32_bf16(false, a, false, b1, (short)0, acc1, false, false);
    }
    __syncthreads();
  }

  const int n = lane & 15;
  const int h = lane >> 4;
  const int col0 = nBase + 32 * nj2 + n;
  const int col1 = col0 + 16;
  const float bz0 = bias[col0];
  const float bz1 = bias[col1];
#pragma unroll
  for (int r = 0; r < 8; ++r) {
    const int grow = rowBase + 16 * mi2 + r + 8 * h;
    const float o0 = acc0[r] + bz0;
    const float o1 = acc1[r] + bz1;
    if (OUT_F32) {
      float* Out = (float*)Outp;
      Out[(size_t)grow * N + col0] = o0;
      Out[(size_t)grow * N + col1] = o1;
    } else {
      __bf16* Out = (__bf16*)Outp;
      Out[(size_t)grow * N + col0] = (__bf16)o0;
      Out[(size_t)grow * N + col1] = (__bf16)o1;
    }
  }
}

// ---------------------------------------------------------------------------
// f32 -> bf16 elementwise (for the raw `value` tensor)
// ---------------------------------------------------------------------------
__global__ void __launch_bounds__(256)
cvt_f32_to_bf16(const float* __restrict__ src, __bf16* __restrict__ dst, size_t n) {
  const size_t i = ((size_t)blockIdx.x * 256 + threadIdx.x) * 8;
  if (i + 8 > n) return;
  *(v8bf*)(dst + i) = cvt8(src + i);
}

// ---------------------------------------------------------------------------
// Flash attention: Ab[b, q, :] = softmax(Qb Kb^T * scale) @ Vb   (bf16 out)
// 256 threads (8 waves). 32 query rows/WG persistent in LDS; key chunks of 32.
// K chunks staged by the Tensor Data Mover (TENSORcnt), double-buffered so the
// DMA of chunk j+1 overlaps softmax+PV of chunk j. V chunk staged pair-packed
// transposed so PV B-fragments are two contiguous 16B LDS loads.
// LDS ~138KB (WGP has 320KB -> 2 WGs/WGP).
// ---------------------------------------------------------------------------
__global__ void __launch_bounds__(256)
attn_kernel(const __bf16* __restrict__ Qb, const __bf16* __restrict__ Kb,
            const __bf16* __restrict__ Vb, __bf16* __restrict__ Ab,
            const float* __restrict__ Tptr, int Bn, int NQ, int NK, int D) {
  __shared__ __bf16   Qf[32 * 512];        // persistent Q tile (32KB)
  __shared__ __bf16   Kf[2][32 * 512];     // double-buffered K chunk (64KB)
  __shared__ unsigned Vt[512 * 16];        // V chunk, {even j, odd j} bf16 pairs (32KB)
  __shared__ float    Ss[2][32 * 32];      // S partials (k-halves) (8KB)
  __shared__ __bf16   Ps[32 * 32];         // probabilities (2KB)
  __shared__ float    mrow[32], lrow[32], arow[32];

  const int tid   = threadIdx.x;
  const int lane  = tid & 31;
  const int wave  = tid >> 5;
  const int nlane = lane & 15;
  const int h     = lane >> 4;
  const int b     = blockIdx.y;
  const int q0    = blockIdx.x * 32;
  const float scale = Tptr[0] * 0.04419417382415922f;  // T / sqrt(512)

  const size_t qrow0 = (size_t)b * NQ + q0;
  const size_t krow0 = (size_t)b * NK;

  // ---- Prologue: kick off TDM for K chunk 0, persist Q tile, init stats
  if (wave == 0)
    tdm_load_k32x512(&Kb[krow0 * D], (unsigned)(uintptr_t)&Kf[0][0]);
  {
    const int r = tid >> 3, c8 = tid & 7;
    const __bf16* qsrc = &Qb[(qrow0 + r) * D + c8 * 64];
#pragma unroll
    for (int i = 0; i < 8; ++i)
      *(v8bf*)&Qf[r * 512 + c8 * 64 + i * 8] = *(const v8bf*)&qsrc[i * 8];
  }
  if (tid < 32) { mrow[tid] = -3.0e38f; lrow[tid] = 0.0f; }

  v8f oacc[2][4];
#pragma unroll
  for (int mi = 0; mi < 2; ++mi)
#pragma unroll
    for (int nt = 0; nt < 4; ++nt) {
      v8f z = {0, 0, 0, 0, 0, 0, 0, 0};
      oacc[mi][nt] = z;
    }

  for (int j0 = 0; j0 < NK; j0 += 32) {
    const int buf = (j0 >> 5) & 1;

    // ---- Stage V chunk (pair-packed transpose), prefetch next V into L2
    {
      const int jp = tid & 15;   // j-pair (j = 2jp, 2jp+1)
      const int dg = tid >> 4;   // d-group of 32
      const v8us* v0 = (const v8us*)&Vb[(krow0 + j0 + 2 * jp) * D + dg * 32];
      const v8us* v1 = (const v8us*)&Vb[(krow0 + j0 + 2 * jp + 1) * D + dg * 32];
#pragma unroll
      for (int i = 0; i < 4; ++i) {
        v8us r0 = v0[i];
        v8us r1 = v1[i];
#pragma unroll
        for (int e = 0; e < 8; ++e) {
          const int d = dg * 32 + i * 8 + e;
          Vt[d * 16 + jp] = (unsigned)r0[e] | ((unsigned)r1[e] << 16);
        }
      }
      if (j0 + 32 < NK)
        __builtin_prefetch(&Vb[(krow0 + j0 + 32 + 2 * jp) * D + dg * 32], 0, 1);
    }

    // ---- Wait for this chunk's TDM, publish; then kick off next chunk's DMA
    __builtin_amdgcn_s_wait_tensorcnt(0);
    __syncthreads();
    if (wave == 0 && j0 + 32 < NK)
      tdm_load_k32x512(&Kb[(krow0 + j0 + 32) * D], (unsigned)(uintptr_t)&Kf[buf ^ 1][0]);

    // ---- S = Q Kchunk^T: all 8 waves, k-split in halves; 8 WMMAs back-to-back
    {
      v8f sacc = {0, 0, 0, 0, 0, 0, 0, 0};
      const int smi = wave & 1;
      const int sni = (wave >> 1) & 1;
      const int kh  = wave >> 2;
      const int kbase = kh * 256;
#pragma unroll
      for (int kk = 0; kk < 256; kk += 32) {
        v16bf a  = frag_a(Qf + kbase + kk, 16 * smi, 512, lane);
        v16bf bb = frag_b(&Kf[buf][0] + kbase + kk, 16 * sni, 512, lane);
        sacc = __builtin_amdgcn_wmma_f32_16x16x32_bf16(false, a, false, bb, (short)0, sacc, false, false);
      }
#pragma unroll
      for (int r = 0; r < 8; ++r)
        Ss[kh][(16 * smi + r + 8 * h) * 32 + 16 * sni + nlane] = sacc[r];
    }
    __syncthreads();

    // ---- Online softmax (8 lanes per row, butterfly reductions)
    {
      const int row = tid >> 3;
      const int c   = tid & 7;
      float s[4];
#pragma unroll
      for (int i = 0; i < 4; ++i)
        s[i] = (Ss[0][row * 32 + c * 4 + i] + Ss[1][row * 32 + c * 4 + i]) * scale;
      float mx = fmaxf(fmaxf(s[0], s[1]), fmaxf(s[2], s[3]));
      for (int off = 1; off < 8; off <<= 1)
        mx = fmaxf(mx, __shfl_xor(mx, off, 32));
      const float mprev = mrow[row];
      const float mnew  = fmaxf(mprev, mx);
      const float al    = __expf(mprev - mnew);
      float p[4], sum = 0.0f;
#pragma unroll
      for (int i = 0; i < 4; ++i) {
        p[i] = __expf(s[i] - mnew);
        Ps[row * 32 + c * 4 + i] = (__bf16)p[i];
        sum += p[i];
      }
      for (int off = 1; off < 8; off <<= 1)
        sum += __shfl_xor(sum, off, 32);
      if (c == 0) {
        lrow[row] = lrow[row] * al + sum;
        mrow[row] = mnew;
        arow[row] = al;
      }
    }
    __syncthreads();

    // ---- Rescale O by alpha, then O += P @ Vchunk (wave owns 64-wide D slice)
    {
      float a_[2][8];
#pragma unroll
      for (int mi = 0; mi < 2; ++mi)
#pragma unroll
        for (int r = 0; r < 8; ++r)
          a_[mi][r] = arow[16 * mi + r + 8 * h];
#pragma unroll
      for (int mi = 0; mi < 2; ++mi)
#pragma unroll
        for (int nt = 0; nt < 4; ++nt)
#pragma unroll
          for (int r = 0; r < 8; ++r)
            oacc[mi][nt][r] *= a_[mi][r];
    }
    {
      v16bf af0 = frag_a(Ps, 0, 32, lane);
      v16bf af1 = frag_a(Ps, 16, 32, lane);
      const int dbase = wave * 64;
#pragma unroll
      for (int nt = 0; nt < 4; ++nt) {
        const int dcol = dbase + 16 * nt + nlane;
        FragBF fb;
        fb.q.lo = *(const uint4*)&Vt[dcol * 16 + 8 * h];
        fb.q.hi = *(const uint4*)&Vt[dcol * 16 + 8 * h + 4];
        oacc[0][nt] = __builtin_amdgcn_wmma_f32_16x16x32_bf16(false, af0, false, fb.v, (short)0, oacc[0][nt], false, false);
        oacc[1][nt] = __builtin_amdgcn_wmma_f32_16x16x32_bf16(false, af1, false, fb.v, (short)0, oacc[1][nt], false, false);
      }
    }
    __syncthreads();
  }

  // ---- Epilogue: divide by running sum, write bf16 attn output
  {
    float linv[2][8];
#pragma unroll
    for (int mi = 0; mi < 2; ++mi)
#pragma unroll
      for (int r = 0; r < 8; ++r)
        linv[mi][r] = 1.0f / lrow[16 * mi + r + 8 * h];
    const int dbase = wave * 64;
#pragma unroll
    for (int mi = 0; mi < 2; ++mi)
#pragma unroll
      for (int nt = 0; nt < 4; ++nt)
#pragma unroll
        for (int r = 0; r < 8; ++r) {
          const size_t grow = qrow0 + 16 * mi + r + 8 * h;
          Ab[grow * D + dbase + 16 * nt + nlane] = (__bf16)(oacc[mi][nt][r] * linv[mi][r]);
        }
  }
}

// ---------------------------------------------------------------------------
extern "C" void kernel_launch(void* const* d_in, const int* in_sizes, int n_in,
                              void* d_out, int out_size, void* d_ws, size_t ws_size,
                              hipStream_t stream) {
  const float* query = (const float*)d_in[0];
  const float* key   = (const float*)d_in[1];
  const float* value = (const float*)d_in[2];
  const float* Wq    = (const float*)d_in[3];
  const float* bq    = (const float*)d_in[4];
  const float* Wk    = (const float*)d_in[5];
  const float* bk    = (const float*)d_in[6];
  // d_in[7]=Wv, d_in[8]=bv: computed-but-unused in the reference; skipped.
  const float* Wo    = (const float*)d_in[9];
  const float* bo    = (const float*)d_in[10];
  const float* Tsc   = (const float*)d_in[11];

  const int Bn = 16, NQ = 2048, NK = 2048, D = 512;
  const int M = Bn * NQ;  // 32768
  const size_t elems = (size_t)M * D;

  // workspace: 4 x 32MB bf16 buffers (Qb, Kb, Vb, Ab) -> 128MB (L2 is 192MB)
  __bf16* Qb = (__bf16*)d_ws;
  __bf16* Kb = Qb + elems;
  __bf16* Vb = Kb + elems;
  __bf16* Ab = Vb + elems;

  dim3 blk(256);
  dim3 gproj(M / 64, D / 64);

  proj_kernel<false, false><<<gproj, blk, 0, stream>>>(query, Wq, bq, Qb, M, D, D);
  proj_kernel<false, false><<<gproj, blk, 0, stream>>>(key,   Wk, bk, Kb, M, D, D);

  const size_t nconv = elems;
  dim3 gconv((unsigned)(nconv / 8 / 256));
  cvt_f32_to_bf16<<<gconv, blk, 0, stream>>>(value, Vb, nconv);

  dim3 gattn(NQ / 32, Bn);
  attn_kernel<<<gattn, blk, 0, stream>>>(Qb, Kb, Vb, Ab, Tsc, Bn, NQ, NK, D);

  proj_kernel<true, true><<<gproj, blk, 0, stream>>>(Ab, Wo, bo, (float*)d_out, M, D, D);
}
